// MambaLayer_88905823027365
// MI455X (gfx1250) — matrix operand order repaired and checked by reading, not compiled, not run on hardware
//
#include <hip/hip_runtime.h>
#include <math.h>

// ---------------- problem constants ----------------
#define D_MODEL 1024
#define D_STATE 16
#define D_CONV  4
#define D_INNER 2048
#define DT_RANK 64
#define B_SZ    2
#define SEQ     4096
#define BL      (B_SZ * SEQ)             // 8192 rows
#define XDBL_W  (DT_RANK + 2 * D_STATE)  // 96

// ---------------- WMMA types ----------------
typedef __bf16 v16bf __attribute__((ext_vector_type(16)));
typedef __bf16 v8bf  __attribute__((ext_vector_type(8)));
typedef float  v8f   __attribute__((ext_vector_type(8)));

// ---------------- tiled split-fp32 (3xBF16) WMMA GEMM ----------------
// C[M,N] = A[M,K] * B[N,K]^T with fp32-class accuracy via hi/lo bf16 split:
//   x ~= hi(x) + lo(x);  x*y ~= hi*hi + hi*lo + lo*hi   (fp32 accumulate)
#define BM 64
#define BN 128
#define KT 32
#define ASTR 40   // padded LDS stride in bf16 elements (row = 80B, 16B aligned)
#define BSTR 40

enum { EPI_NONE = 0, EPI_BIAS_SOFTPLUS = 1 };

#define SHUF16(x, y) __builtin_shufflevector((x), (y), 0, 1, 2, 3, 4, 5, 6, 7, \
                                             8, 9, 10, 11, 12, 13, 14, 15)

template <int EPI>
__launch_bounds__(256)
__global__ void wmma_gemm_split_bf16(const float* __restrict__ A,
                                     const float* __restrict__ Bm,
                                     const float* __restrict__ bias,
                                     float* __restrict__ C,
                                     int M, int N, int K,
                                     int lda, int ldb, int ldc) {
    __shared__ __bf16 lAhi[2][BM * ASTR];
    __shared__ __bf16 lAlo[2][BM * ASTR];
    __shared__ __bf16 lBhi[2][BN * BSTR];
    __shared__ __bf16 lBlo[2][BN * BSTR];

    const int tid  = threadIdx.x;
    const int lane = tid & 31;
    const int wave = tid >> 5;       // 0..7
    const int wm   = wave & 3;       // 0..3 : 16-row sub-tile
    const int wn   = wave >> 2;      // 0..1 : 64-col sub-tile
    const int m15  = lane & 15;
    const int half = lane >> 4;      // 0/1

    const int rowBase = blockIdx.y * BM;
    const int colBase = blockIdx.x * BN;

    v8f acc[4] = {v8f{}, v8f{}, v8f{}, v8f{}};

    // staging map: A tile 64x32 floats (8/thread), B tile 128x32 (16/thread)
    const int rowA = tid >> 2;          // 0..63
    const int kA   = (tid & 3) * 8;     // 0,8,16,24
    const int rowB = tid >> 1;          // 0..127
    const int kB   = (tid & 1) * 16;    // 0,16
    const int bcol = colBase + rowB;

    float ra[8];
    float rb[16];

    // global tile fetch into registers (coalesced float4)
    auto gload = [&](int k0) {
        const float* gp = &A[(size_t)(rowBase + rowA) * lda + k0 + kA];
        const float4 a0 = *reinterpret_cast<const float4*>(gp);
        const float4 a1 = *reinterpret_cast<const float4*>(gp + 4);
        ra[0]=a0.x; ra[1]=a0.y; ra[2]=a0.z; ra[3]=a0.w;
        ra[4]=a1.x; ra[5]=a1.y; ra[6]=a1.z; ra[7]=a1.w;
        if (bcol < N) {
            const float* bp = &Bm[(size_t)bcol * ldb + k0 + kB];
            const float4 b0 = *reinterpret_cast<const float4*>(bp);
            const float4 b1 = *reinterpret_cast<const float4*>(bp + 4);
            const float4 b2 = *reinterpret_cast<const float4*>(bp + 8);
            const float4 b3 = *reinterpret_cast<const float4*>(bp + 12);
            rb[0]=b0.x; rb[1]=b0.y; rb[2]=b0.z; rb[3]=b0.w;
            rb[4]=b1.x; rb[5]=b1.y; rb[6]=b1.z; rb[7]=b1.w;
            rb[8]=b2.x; rb[9]=b2.y; rb[10]=b2.z; rb[11]=b2.w;
            rb[12]=b3.x; rb[13]=b3.y; rb[14]=b3.z; rb[15]=b3.w;
        } else {
#pragma unroll
            for (int i = 0; i < 16; ++i) rb[i] = 0.f;
        }
    };

    gload(0);
    int buf = 0;

    for (int k0 = 0; k0 < K; k0 += KT) {
        // ---- convert regs -> (hi,lo) bf16, vector LDS stores ----
        {
            v8bf hi, lo;
#pragma unroll
            for (int i = 0; i < 8; ++i) {
                const __bf16 h = (__bf16)ra[i];
                hi[i] = h;
                lo[i] = (__bf16)(ra[i] - (float)h);
            }
            *(v8bf*)&lAhi[buf][rowA * ASTR + kA] = hi;
            *(v8bf*)&lAlo[buf][rowA * ASTR + kA] = lo;
        }
#pragma unroll
        for (int q = 0; q < 2; ++q) {
            v8bf hi, lo;
#pragma unroll
            for (int i = 0; i < 8; ++i) {
                const float v = rb[q * 8 + i];
                const __bf16 h = (__bf16)v;
                hi[i] = h;
                lo[i] = (__bf16)(v - (float)h);
            }
            *(v8bf*)&lBhi[buf][rowB * BSTR + kB + q * 8] = hi;
            *(v8bf*)&lBlo[buf][rowB * BSTR + kB + q * 8] = lo;
        }
        __syncthreads();

        // ---- prefetch next tile while computing this one ----
        if (k0 + KT < K) gload(k0 + KT);

        // ---- A fragments (16-bit A 16x32 ISA layout) ----
        // lane half=0: K = 0..7 and 16..23 ; half=1: K = 8..15 and 24..31
        const int abase = (wm * 16 + m15) * ASTR + half * 8;
        const v16bf ahi = SHUF16(*(const v8bf*)&lAhi[buf][abase],
                                 *(const v8bf*)&lAhi[buf][abase + 16]);
        const v16bf alo = SHUF16(*(const v8bf*)&lAlo[buf][abase],
                                 *(const v8bf*)&lAlo[buf][abase + 16]);

        // ---- B fragments + 3 WMMAs per 16-col strip ----
#pragma unroll
        for (int j = 0; j < 4; ++j) {
            const int bbase = (wn * 64 + j * 16 + m15) * BSTR + half * 8;
            const v16bf bhi = SHUF16(*(const v8bf*)&lBhi[buf][bbase],
                                     *(const v8bf*)&lBhi[buf][bbase + 16]);
            const v16bf blo = SHUF16(*(const v8bf*)&lBlo[buf][bbase],
                                     *(const v8bf*)&lBlo[buf][bbase + 16]);
            acc[j] = __builtin_amdgcn_wmma_f32_16x16x32_bf16(
                         false, ahi, false, bhi, (short)0, acc[j], false, false);
            acc[j] = __builtin_amdgcn_wmma_f32_16x16x32_bf16(
                         false, ahi, false, blo, (short)0, acc[j], false, false);
            acc[j] = __builtin_amdgcn_wmma_f32_16x16x32_bf16(
                         false, alo, false, bhi, (short)0, acc[j], false, false);
        }
        buf ^= 1;
    }

    // ---- store: D layout -> VGPR r holds row (r + 8*half), col = lane&15 ----
    const int crow = rowBase + wm * 16 + half * 8;
#pragma unroll
    for (int j = 0; j < 4; ++j) {
        const int ccol = colBase + wn * 64 + j * 16 + m15;
        if (ccol < N) {
#pragma unroll
            for (int r = 0; r < 8; ++r) {
                float v = acc[j][r];
                if (EPI == EPI_BIAS_SOFTPLUS) {
                    v += bias[ccol];
                    v = (v > 20.f) ? v : logf(1.f + __expf(v));
                }
                C[(size_t)(crow + r) * ldc + ccol] = v;
            }
        }
    }
}

// ---------------- causal depthwise conv1d + bias + SiLU ----------------
__launch_bounds__(256)
__global__ void conv_silu_kernel(const float* __restrict__ xz,
                                 const float* __restrict__ w,
                                 const float* __restrict__ cb,
                                 float* __restrict__ x_conv) {
    const size_t i = (size_t)blockIdx.x * 256 + threadIdx.x; // over BL*D_INNER
    const int d = (int)(i % D_INNER);
    const size_t bl = i / D_INNER;  // b*SEQ + l
    const int l = (int)(bl % SEQ);

    float s = cb[d];
#pragma unroll
    for (int k = 0; k < D_CONV; ++k) {
        const int dl = l + k - (D_CONV - 1);
        if (dl >= 0)
            s += w[d * D_CONV + k] *
                 xz[(bl + (size_t)(k - (D_CONV - 1))) * (size_t)(2 * D_INNER) + d];
    }
    x_conv[i] = s / (1.f + __expf(-s));   // silu(s)
}

// ---------------- selective scan: 16 lanes per channel (one per state) ----------------
__launch_bounds__(256)
__global__ void mamba_scan_kernel(const float* __restrict__ xz,
                                  const float* __restrict__ x_conv,
                                  const float* __restrict__ x_dbl,
                                  const float* __restrict__ dt,
                                  const float* __restrict__ A_log,
                                  const float* __restrict__ Dp,
                                  float* __restrict__ y) {
    const int gid = blockIdx.x * blockDim.x + threadIdx.x;
    const int c = gid >> 4;       // channel in [0, B*D_INNER)
    const int n = gid & 15;       // state index
    const int b = c / D_INNER;
    const int d = c % D_INNER;

    const float Aval = -__expf(A_log[d * D_STATE + n]);
    const float Dval = Dp[d];

    const float* xc_p = x_conv + (size_t)b * SEQ * D_INNER + d;
    const float* dt_p = dt     + (size_t)b * SEQ * D_INNER + d;
    const float* bd_p = x_dbl  + (size_t)b * SEQ * XDBL_W;
    const float* z_p  = xz     + (size_t)b * SEQ * (2 * D_INNER) + D_INNER + d;
    float*       y_p  = y      + (size_t)b * SEQ * D_INNER + d;

    float h = 0.f;
    for (int t = 0; t < SEQ; ++t) {
        const int tp = (t + 4 < SEQ) ? t + 4 : t;
        __builtin_prefetch(&xc_p[(size_t)tp * D_INNER], 0, 1);
        __builtin_prefetch(&bd_p[(size_t)tp * XDBL_W + DT_RANK + n], 0, 1);

        const float xv  = xc_p[(size_t)t * D_INNER];
        const float dtv = dt_p[(size_t)t * D_INNER];
        const float Bv  = bd_p[(size_t)t * XDBL_W + DT_RANK + n];
        const float Cv  = bd_p[(size_t)t * XDBL_W + DT_RANK + D_STATE + n];

        h = __expf(dtv * Aval) * h + (dtv * xv) * Bv;

        float p = h * Cv;   // reduce over 16 states (stays inside 16-lane group)
        p += __shfl_xor(p, 1, 32);
        p += __shfl_xor(p, 2, 32);
        p += __shfl_xor(p, 4, 32);
        p += __shfl_xor(p, 8, 32);

        if (n == 0) {
            const float zv = z_p[(size_t)t * 2 * D_INNER];
            const float sz = zv / (1.f + __expf(-zv));  // silu(z)
            y_p[(size_t)t * D_INNER] = (p + Dval * xv) * sz;
        }
    }
}

// ---------------- host-side launcher ----------------
extern "C" void kernel_launch(void* const* d_in, const int* in_sizes, int n_in,
                              void* d_out, int out_size, void* d_ws, size_t ws_size,
                              hipStream_t stream) {
    const float* x      = (const float*)d_in[0];
    const float* W_in   = (const float*)d_in[1];
    const float* conv_w = (const float*)d_in[2];
    const float* conv_b = (const float*)d_in[3];
    const float* W_x    = (const float*)d_in[4];
    const float* W_dt   = (const float*)d_in[5];
    const float* b_dt   = (const float*)d_in[6];
    const float* A_log  = (const float*)d_in[7];
    const float* Dvec   = (const float*)d_in[8];
    const float* W_out  = (const float*)d_in[9];
    float* out = (float*)d_out;

    // workspace carving (float counts; all 16B aligned)
    float* ws = (float*)d_ws;
    float* xz     = ws;                                 // [BL, 2*D_INNER]
    float* x_conv = xz     + (size_t)BL * 2 * D_INNER;  // [BL, D_INNER]
    float* x_dbl  = x_conv + (size_t)BL * D_INNER;      // [BL, 96]
    float* dt     = x_dbl  + (size_t)BL * XDBL_W;       // [BL, D_INNER]
    float* y      = dt     + (size_t)BL * D_INNER;      // [BL, D_INNER]

    const dim3 blk(256);

    // 1) xz = x @ W_in^T     (M=8192, N=4096, K=1024)
    wmma_gemm_split_bf16<EPI_NONE><<<dim3((2 * D_INNER) / BN, BL / BM), blk, 0, stream>>>(
        x, W_in, nullptr, xz, BL, 2 * D_INNER, D_MODEL, D_MODEL, D_MODEL, 2 * D_INNER);

    // 2) depthwise causal conv + bias + SiLU
    conv_silu_kernel<<<((size_t)BL * D_INNER) / 256, blk, 0, stream>>>(
        xz, conv_w, conv_b, x_conv);

    // 3) x_dbl = x_conv @ W_x^T   (M=8192, N=96, K=2048)
    wmma_gemm_split_bf16<EPI_NONE><<<dim3((XDBL_W + BN - 1) / BN, BL / BM), blk, 0, stream>>>(
        x_conv, W_x, nullptr, x_dbl, BL, XDBL_W, D_INNER, D_INNER, D_INNER, XDBL_W);

    // 4) dt = softplus(dt_low @ W_dt^T + b_dt)   (M=8192, N=2048, K=64)
    wmma_gemm_split_bf16<EPI_BIAS_SOFTPLUS><<<dim3(D_INNER / BN, BL / BM), blk, 0, stream>>>(
        x_dbl, W_dt, b_dt, dt, BL, D_INNER, DT_RANK, XDBL_W, DT_RANK, D_INNER);

    // 5) selective scan (+ D*x skip, * silu(z))
    mamba_scan_kernel<<<(B_SZ * D_INNER * 16) / 256, blk, 0, stream>>>(
        xz, x_conv, x_dbl, dt, A_log, Dvec, y);

    // 6) out = y @ W_out^T    (M=8192, N=1024, K=2048)
    wmma_gemm_split_bf16<EPI_NONE><<<dim3(D_MODEL / BN, BL / BM), blk, 0, stream>>>(
        y, W_out, nullptr, out, BL, D_MODEL, D_INNER, D_INNER, D_INNER, D_MODEL);
}
